// ChebyKANLinear_1589137900066
// MI455X (gfx1250) — matrix operand order repaired and checked by reading, not compiled, not run on hardware
//
#include <hip/hip_runtime.h>
#include <hip/hip_bf16.h>
#include <stdint.h>

// Problem constants (from reference)
#define BQ    16384
#define CIN   512
#define COUT  512
#define DEG1  9                  // degree+1
#define KTOT  (CIN * DEG1)       // 4608  (GEMM K)

// Tiling
#define BM    128
#define BN    128
#define CI    32                 // input channels per outer chunk
#define KT    (CI * DEG1)        // 288 = 9 WMMA k-steps of 32
#define KTP   (KT + 8)           // padded LDS row stride (bf16 elems)
#define NTHREADS 256

#define AS1 __attribute__((address_space(1)))
#define AS3 __attribute__((address_space(3)))

#if defined(__has_builtin)
#  if __has_builtin(__builtin_amdgcn_global_load_async_to_lds_b128)
#    define HAVE_ASYNC_COPY 1
#  endif
#endif
#ifndef HAVE_ASYNC_COPY
#  define HAVE_ASYNC_COPY 0
#endif

typedef __attribute__((ext_vector_type(16))) __bf16 v16bf;
typedef __attribute__((ext_vector_type(8)))  __bf16 bf16x8;
typedef __attribute__((ext_vector_type(2)))  __bf16 bf16x2;
typedef __attribute__((ext_vector_type(8)))  float  v8f;
typedef __attribute__((ext_vector_type(4)))  int    v4i;

// Hardware transcendental tanh (gfx1250 V_TANH_F32). §7.4: one independent
// op / V_NOP needed after a TRANS op before its result is consumed.
__device__ __forceinline__ float hw_tanh(float v) {
  float r;
  asm("v_tanh_f32 %0, %1\n\tv_nop" : "=v"(r) : "v"(v));
  return r;
}

__device__ __forceinline__ void wait_async_zero() {
#if defined(__has_builtin) && __has_builtin(__builtin_amdgcn_s_wait_asynccnt)
  __builtin_amdgcn_s_wait_asynccnt(0);
#else
  asm volatile("s_wait_asynccnt 0x0" ::: "memory");
#endif
}

// K-index bijection shared by A and B: k = (i/2)*18 + 2*d + (i%2).
// Lets the basis writer emit 9 contiguous dwords (two interleaved channels).

// ---------------------------------------------------------------------------
// Repack coeffs (c_in, c_out, 9) f32 -> Bpack[n][k] bf16 (K-contiguous rows).
// 2.36M elements; runs once per launch, output lives in L2 afterwards.
// ---------------------------------------------------------------------------
__global__ __launch_bounds__(NTHREADS)
void cheby_repack_coeffs(const float* __restrict__ coeffs,
                         __hip_bfloat16* __restrict__ bp) {
  const int idx = blockIdx.x * NTHREADS + threadIdx.x;
  if (idx >= COUT * KTOT) return;
  const int n  = idx / KTOT;
  const int kk = idx - n * KTOT;
  const int i  = ((kk / 18) << 1) + (kk & 1);
  const int d  = (kk % 18) >> 1;
  bp[idx] = __float2bfloat16(coeffs[(size_t)i * KTOT + n * DEG1 + d]);
}

// ---------------------------------------------------------------------------
// Fused basis + GEMM. Each workgroup: 128x128 output tile, 8 waves in a
// 2(M) x 4(N) grid, each wave owns 4x2 WMMA 16x16 tiles (64 f32 acc VGPRs).
// Per chunk: (1) async-DMA the B tile into LDS (ASYNCcnt path), overlapping
//            (2) tanh + Chebyshev-recurrence basis tile written to LDS,
//            (3) 9 WMMA k-steps; next chunk's x slice prefetched meanwhile.
// ---------------------------------------------------------------------------
template <bool PACKED>
__global__ __launch_bounds__(NTHREADS)
void cheby_fused(const float* __restrict__ x, const void* __restrict__ bsrc,
                 float* __restrict__ out) {
  __shared__ __align__(16) __bf16 sA[BM * KTP];   // basis tile, row-major [m][k]
  __shared__ __align__(16) __bf16 sB[BN * KTP];   // B tile, [n][k] (col of B contiguous)

  const int tid  = threadIdx.x;
  const int lane = tid & 31;
  const int wave = tid >> 5;
  const int wm   = wave & 1;        // 0..1 -> 64-row half
  const int wn   = wave >> 1;       // 0..3 -> 32-col quarter
  const int l16  = lane & 15;
  const int lh   = lane >> 4;       // lane half (ISA K-striping)
  const int m0   = blockIdx.y * BM;
  const int n0   = blockIdx.x * BN;

  v8f acc[4][2];
#pragma unroll
  for (int mt = 0; mt < 4; ++mt)
#pragma unroll
    for (int nt = 0; nt < 2; ++nt)
      acc[mt][nt] = (v8f){0.f, 0.f, 0.f, 0.f, 0.f, 0.f, 0.f, 0.f};

  for (int ci0 = 0; ci0 < CIN; ci0 += CI) {
    // -------- Phase A: B coefficient tile -> sB (128 cols x 288 k) ---------
    if (PACKED) {
      const __bf16* bp = (const __bf16*)bsrc;
#if HAVE_ASYNC_COPY
      // Async DMA: no VGPR staging, tracked on ASYNCcnt; overlaps Phase B.
#pragma unroll
      for (int r = 0; r < 18; ++r) {
        const int c  = r * NTHREADS + tid;  // 0..4607 16B-chunks
        const int n  = c / 36;
        const int kc = (c - n * 36) * 8;
        const __bf16* g = bp + (size_t)(n0 + n) * KTOT + ci0 * DEG1 + kc;
        __builtin_amdgcn_global_load_async_to_lds_b128(
            (AS1 v4i*)(AS1 void*)g,
            (AS3 v4i*)(AS3 void*)(sB + n * KTP + kc), 0, 0);
      }
#else
#pragma unroll
      for (int r = 0; r < 18; ++r) {
        const int c  = r * NTHREADS + tid;
        const int n  = c / 36;
        const int kc = (c - n * 36) * 8;
        *(bf16x8*)(sB + n * KTP + kc) =
            *(const bf16x8*)(bp + (size_t)(n0 + n) * KTOT + ci0 * DEG1 + kc);
      }
#endif
    } else {
      // Fallback: gather f32 coeffs directly (no workspace needed).
      const float* cf = (const float*)bsrc;
      for (int r = 0; r < (BN * KT) / NTHREADS; ++r) {
        const int idx = r * NTHREADS + tid;
        const int n = idx / KT;
        const int k = idx - n * KT;
        const int il = ((k / 18) << 1) + (k & 1);
        const int d  = (k % 18) >> 1;
        sB[n * KTP + k] =
            (__bf16)cf[(size_t)(ci0 + il) * KTOT + (n0 + n) * DEG1 + d];
      }
    }

    // -------- Phase B: Chebyshev basis tile -> sA (128 rows x 288 k) -------
    // 2048 (row, channel-pair) tasks; each: float2 load, 4 tanh, recurrence,
    // 9 packed-bf16x2 dword stores (contiguous 36B).
#pragma unroll
    for (int r = 0; r < 8; ++r) {
      const int p  = r * NTHREADS + tid;   // 0..2047
      const int m  = p >> 4;               // 0..127
      const int ip = p & 15;               // channel pair 0..15
      const float2 xv =
          *(const float2*)(x + (size_t)(m0 + m) * CIN + ci0 + ip * 2);
      const float u0 = hw_tanh(hw_tanh(xv.x));
      const float u1 = hw_tanh(hw_tanh(xv.y));
      bf16x2* dst = (bf16x2*)(sA + m * KTP + ip * 18);
      float pa0 = 1.0f, pb0 = 1.0f, pa1 = u0, pb1 = u1;
      bf16x2 w;
      w.x = (__bf16)1.0f; w.y = (__bf16)1.0f; dst[0] = w;
      w.x = (__bf16)u0;   w.y = (__bf16)u1;   dst[1] = w;
      const float t0 = 2.0f * u0, t1 = 2.0f * u1;
#pragma unroll
      for (int d = 2; d < DEG1; ++d) {
        const float na = __builtin_fmaf(t0, pa1, -pa0);
        const float nb = __builtin_fmaf(t1, pb1, -pb0);
        pa0 = pa1; pb0 = pb1; pa1 = na; pb1 = nb;
        w.x = (__bf16)na; w.y = (__bf16)nb; dst[d] = w;
      }
    }

#if HAVE_ASYNC_COPY
    if (PACKED) wait_async_zero();   // B tile DMA complete before barrier
#endif
    __syncthreads();

    // -------- Phase C: 9 WMMA k-steps over the chunk -----------------------
    // Prefetch next chunk's x slice into cache while the WMMAs run.
    if (ci0 + CI < CIN) {
#pragma unroll
      for (int r = 0; r < 8; ++r) {
        const int p  = r * NTHREADS + tid;
        const int m  = p >> 4;
        const int ip = p & 15;
        __builtin_prefetch(x + (size_t)(m0 + m) * CIN + (ci0 + CI) + ip * 2,
                           0, 1);
      }
    }
#pragma unroll
    for (int ks = 0; ks < DEG1; ++ks) {
      const int kb = ks * 32;
      v16bf af[4];
      v16bf bfr[2];
      // A fragment (16x32 bf16): lane l -> row M=l%16; VGPR0-3 = K base +
      // (l/16)*8 .. +7 (16B), VGPR4-7 = +16 (second 16B).  [ISA 7.12.2]
#pragma unroll
      for (int mt = 0; mt < 4; ++mt) {
        const int row = wm * 64 + mt * 16 + l16;
        const __bf16* pA = sA + row * KTP + kb + lh * 8;
        const bf16x8 lo = *(const bf16x8*)pA;
        const bf16x8 hi = *(const bf16x8*)(pA + 16);
        af[mt] = __builtin_shufflevector(lo, hi, 0, 1, 2, 3, 4, 5, 6, 7, 8, 9,
                                         10, 11, 12, 13, 14, 15);
      }
      // B fragment (32x16 bf16): lane l -> col N=l%16, K = (l/16)*8 + 0..7
      // and +16 (same order as A) -> two b128 loads from the [n][k] tile.
#pragma unroll
      for (int nt = 0; nt < 2; ++nt) {
        const int col = wn * 32 + nt * 16 + l16;
        const __bf16* pB = sB + col * KTP + kb + lh * 8;
        const bf16x8 lo = *(const bf16x8*)pB;
        const bf16x8 hi = *(const bf16x8*)(pB + 16);
        bfr[nt] = __builtin_shufflevector(lo, hi, 0, 1, 2, 3, 4, 5, 6, 7, 8, 9,
                                          10, 11, 12, 13, 14, 15);
      }
#pragma unroll
      for (int mt = 0; mt < 4; ++mt)
#pragma unroll
        for (int nt = 0; nt < 2; ++nt)
          acc[mt][nt] = __builtin_amdgcn_wmma_f32_16x16x32_bf16(
              false, af[mt], false, bfr[nt], (short)0, acc[mt][nt], false,
              false);
    }
    __syncthreads();   // tile consumed; safe to refill next chunk
  }

  // -------- Epilogue: C/D layout -> f32 rows (VGPR r: M = r + (l/16)*8) ----
#pragma unroll
  for (int mt = 0; mt < 4; ++mt)
#pragma unroll
    for (int nt = 0; nt < 2; ++nt) {
      const int n = n0 + wn * 32 + nt * 16 + l16;
#pragma unroll
      for (int r = 0; r < 8; ++r) {
        const int m = m0 + wm * 64 + mt * 16 + lh * 8 + r;
        out[(size_t)m * COUT + n] = acc[mt][nt][r];
      }
    }
}

// ---------------------------------------------------------------------------
extern "C" void kernel_launch(void* const* d_in, const int* in_sizes, int n_in,
                              void* d_out, int out_size, void* d_ws,
                              size_t ws_size, hipStream_t stream) {
  const float* x      = (const float*)d_in[0];       // (16384, 512) f32
  const float* coeffs = (const float*)d_in[1];       // (512, 512, 9) f32
  float* out          = (float*)d_out;               // (16384, 512) f32

  dim3 grid(COUT / BN, BQ / BM);                     // (4, 128)
  const size_t packBytes = (size_t)COUT * KTOT * 2;  // 4.7 MB bf16 panel

  if (ws_size >= packBytes) {
    __hip_bfloat16* bp = (__hip_bfloat16*)d_ws;
    const int total = COUT * KTOT;
    cheby_repack_coeffs<<<(total + NTHREADS - 1) / NTHREADS, NTHREADS, 0,
                          stream>>>(coeffs, bp);
    cheby_fused<true><<<grid, NTHREADS, 0, stream>>>(x, (const void*)bp, out);
  } else {
    cheby_fused<false><<<grid, NTHREADS, 0, stream>>>(x, (const void*)coeffs,
                                                      out);
  }
}
